// BiLSTM_9972914061892
// MI455X (gfx1250) — compile-verified, matrix operand order checked by reading, not run on hardware
//
#include <hip/hip_runtime.h>
#include <hip/hip_bf16.h>

// ---------------------------------------------------------------------------
// BiLSTM tagger for MI455X (gfx1250, wave32, WMMA bf16 16x16x32, f32 accum).
// Pipeline:
//   1. pack W/U/Wd (f32 -> bf16) into WMMA B-fragment layout
//   2. embedding gather -> bf16 A-fragment layout
//   3. xz = x @ W (fwd + bwd) via WMMA -> f32 [B*T, 4H]
//   4. recurrent scan: 16 batch rows / workgroup, 16 waves, gates fused
//      on-accumulator; h kept in LDS in A-fragment layout; next-step xz
//      prefetched with global_prefetch during the WMMA phase
//   5. dense [B*T,2H]@[2H,17] via WMMA + softmax
// ---------------------------------------------------------------------------

typedef __attribute__((ext_vector_type(16))) __bf16 v16bf;
typedef __attribute__((ext_vector_type(8)))  float  v8f;

#define WMMA_BF16(a, b, c) \
  __builtin_amdgcn_wmma_f32_16x16x32_bf16(false, (a), false, (b), (short)0, (c), false, false)

static constexpr int BB    = 256;   // batch
static constexpr int TT    = 128;   // seq len
static constexpr int EE    = 256;   // embed dim
static constexpr int HH    = 256;   // hidden
static constexpr int G4H   = 1024;  // 4*H
static constexpr int NTAGS = 17;
static constexpr int ROWS  = BB * TT;       // 32768
static constexpr int MT    = ROWS / 16;     // 2048 M-tiles

// --- K permutation for 16-bit A/B fragments (ISA 7.12.2, 16-bit A 16x32) ----
// lanes 0-15 hold K = {0..7, 16..23}; lanes 16-31 hold K = {8..15, 24..31}
__device__ __forceinline__ int kperm(int g, int s) {
  int vg = s >> 1, p = s & 1;
  int kb = (vg < 4) ? (vg * 2 + p) : (16 + (vg - 4) * 2 + p);
  return g * 8 + kb;
}
__device__ __forceinline__ void kinv(int kk, int& g, int& s) {
  g = ((kk & 15) >= 8) ? 1 : 0;
  int kb = kk - 8 * g;
  s = (kb < 8) ? kb : (kb - 8);
}
__device__ __forceinline__ float sigm(float x) { return 1.f / (1.f + __expf(-x)); }

// ---------------------------------------------------------------------------
// Pack f32 weight matrix [K rows x N cols] into bf16 B-fragment layout
// dst[kt][nt][lane][slot], zero-padding out-of-range (for Wd: N=17 -> 32).
// ---------------------------------------------------------------------------
__global__ void pack_b_kernel(const float* __restrict__ W, __bf16* __restrict__ dst,
                              int K, int N, int KT, int NT) {
  int idx = blockIdx.x * 256 + threadIdx.x;
  int total = KT * NT * 512;
  if (idx >= total) return;
  int slot = idx & 15;
  int lane = (idx >> 4) & 31;
  int nt   = (idx >> 9) % NT;
  int kt   = idx / (512 * NT);
  int n = nt * 16 + (lane & 15);
  int g = lane >> 4;
  int k = kt * 32 + kperm(g, slot);
  float v = (k < K && n < N) ? W[(size_t)k * N + n] : 0.f;
  dst[idx] = (__bf16)v;
}

// ---------------------------------------------------------------------------
// Embedding gather -> bf16 A-fragment layout xa[mtile][kt(8)][lane][slot]
// ---------------------------------------------------------------------------
__global__ void gather_pack_kernel(const int* __restrict__ tokens,
                                   const float* __restrict__ emb,
                                   __bf16* __restrict__ xa) {
  int row = blockIdx.x;          // b*T + t
  int e   = threadIdx.x;         // 0..255
  int tok = tokens[row];
  float v = emb[(size_t)tok * EE + e];
  int mtile = row >> 4, m = row & 15;
  int kt = e >> 5, kk = e & 31;
  int g, s; kinv(kk, g, s);
  int lane = g * 16 + m;
  xa[(((size_t)mtile * 8 + kt) * 32 + lane) * 16 + s] = (__bf16)v;
}

// ---------------------------------------------------------------------------
// Input projection: out[32768,1024] f32 = xa(bf16 A-frags) @ Bpack(bf16)
// grid = 2048 blocks x 256 threads; wave w -> n-tiles [8w, 8w+8)
// ---------------------------------------------------------------------------
__global__ void proj_gemm_kernel(const __bf16* __restrict__ Apack,
                                 const __bf16* __restrict__ Bpack,
                                 float* __restrict__ out) {
  int mtile = blockIdx.x;
  int wave = threadIdx.x >> 5, lane = threadIdx.x & 31;
  const v8f vz = {0.f, 0.f, 0.f, 0.f, 0.f, 0.f, 0.f, 0.f};
  v8f acc[8];
#pragma unroll
  for (int j = 0; j < 8; ++j) acc[j] = vz;

  const __bf16* Ab = Apack + (size_t)mtile * 8 * 512;
#pragma unroll
  for (int kt = 0; kt < 8; ++kt) {
    v16bf a = *(const v16bf*)(Ab + ((size_t)kt * 32 + lane) * 16);
#pragma unroll
    for (int j = 0; j < 8; ++j) {
      int nt = wave * 8 + j;
      v16bf b = *(const v16bf*)(Bpack + (((size_t)kt * 64 + nt) * 32 + lane) * 16);
      acc[j] = WMMA_BF16(a, b, acc[j]);
    }
  }
  int col = lane & 15, ro = (lane >> 4) * 8;
#pragma unroll
  for (int j = 0; j < 8; ++j) {
#pragma unroll
    for (int r = 0; r < 8; ++r) {
      out[((size_t)mtile * 16 + ro + r) * G4H + (wave * 8 + j) * 16 + col] = acc[j][r];
    }
  }
}

// ---------------------------------------------------------------------------
// Recurrent scan. grid = (16, 2): x -> 16-row batch group, y -> direction.
// 512 threads = 16 waves. Wave w owns h-columns [16w, 16w+16) of ALL four
// gates (tiles nt = {w, 16+w, 32+w, 48+w}), so the cell update runs entirely
// on its own accumulators in C-fragment layout -> only 32 WMMAs per wave per
// step on the serial critical path. New h is written to LDS pre-swizzled into
// A-fragment layout for the next step, and scattered to global hpack (A-frag
// layout of [32768 x 512]) for the dense layer. The next timestep's xz lines
// are warmed with global_prefetch during the WMMA phase.
// ---------------------------------------------------------------------------
__global__ void __launch_bounds__(512)
scan_kernel(const float* __restrict__ xzf, const float* __restrict__ xzb,
            const float* __restrict__ bf_, const float* __restrict__ bb_,
            const __bf16* __restrict__ Uf, const __bf16* __restrict__ Ub,
            __bf16* __restrict__ hpack) {
  const int dir = blockIdx.y;
  const int b0  = blockIdx.x * 16;
  const float*  xz   = dir ? xzb : xzf;
  const float*  bias = dir ? bb_ : bf_;
  const __bf16* U    = dir ? Ub  : Uf;

  __shared__ __align__(32) __bf16 hsw[8][32][16];  // h in A-fragment layout (8 KB)

  const int tid = threadIdx.x;
  const int wave = tid >> 5, lane = tid & 31;
  const int col = lane & 15, ro = (lane >> 4) * 8;
  const int nc = wave * 16 + col;  // this thread's h column (0..255)

  for (int i = tid; i < 8 * 32 * 16; i += 512) ((__bf16*)hsw)[i] = (__bf16)0.f;
  float cst[8];
#pragma unroll
  for (int i = 0; i < 8; ++i) cst[i] = 0.f;

  // hoisted per-thread constants
  const float bi0 = bias[nc];
  const float bi1 = bias[HH + nc];
  const float bi2 = bias[2 * HH + nc];
  const float bi3 = bias[3 * HH + nc];
  int g2, s2;  kinv(nc & 31, g2, s2);          // hsw write swizzle
  const int kt2 = nc >> 5;
  const int kd  = dir * HH + nc;               // hpack K index
  int g3, s3;  kinv(kd & 31, g3, s3);
  const int kt3 = kd >> 5;

  __syncthreads();

  const v8f vz = {0.f, 0.f, 0.f, 0.f, 0.f, 0.f, 0.f, 0.f};

  for (int t = 0; t < TT; ++t) {
    const int ttv = dir ? (TT - 1 - t) : t;
    const int tn  = (t + 1 < TT) ? (t + 1) : t;
    const int ttn = dir ? (TT - 1 - tn) : tn;

    // warm next step's xz cachelines (2 x 64B lines per thread, full EXEC):
    // line id = col + 16*u covers (gate p, local row r) for this wave's cols
#pragma unroll
    for (int u = 0; u < 2; ++u) {
      int id = col + u * 16;                   // 0..31
      int p = id >> 3, r = id & 7;
      const float* pfa =
          xz + ((size_t)(b0 + ro + r) * TT + ttn) * (size_t)G4H + p * HH + wave * 16;
      __builtin_prefetch(pfa, 0, 3);
    }

    v8f acc[4];  // gate p tile, nt = p*16 + wave
#pragma unroll
    for (int p = 0; p < 4; ++p) acc[p] = vz;

#pragma unroll
    for (int kt = 0; kt < 8; ++kt) {
      v16bf a = *(const v16bf*)&hsw[kt][lane][0];
#pragma unroll
      for (int p = 0; p < 4; ++p) {
        int nt = p * 16 + wave;
        v16bf b = *(const v16bf*)(U + (((size_t)kt * 64 + nt) * 32 + lane) * 16);
        acc[p] = WMMA_BF16(a, b, acc[p]);
      }
    }
    __syncthreads();  // all A-frag reads of hsw done before we overwrite it

#pragma unroll
    for (int r = 0; r < 8; ++r) {
      const int rl  = ro + r;              // local row 0..15
      const int row = b0 + rl;             // batch index
      const size_t base = ((size_t)row * TT + ttv) * (size_t)G4H;
      float zi = acc[0][r] + xz[base + nc]          + bi0;
      float zf = acc[1][r] + xz[base + HH + nc]     + bi1;
      float zg = acc[2][r] + xz[base + 2 * HH + nc] + bi2;
      float zo = acc[3][r] + xz[base + 3 * HH + nc] + bi3;
      float ig = sigm(zi), fg = sigm(zf), gg = tanhf(zg), og = sigm(zo);
      float cn = fg * cst[r] + ig * gg;
      cst[r] = cn;
      float h = og * tanhf(cn);

      // LDS: A-fragment layout for next step's WMMA
      hsw[kt2][g2 * 16 + rl][s2] = (__bf16)h;

      // Global: A-fragment layout of the [32768 x 512] dense input
      int rowM = row * TT + ttv;
      int mt2 = rowM >> 4, m2 = rowM & 15;
      hpack[(((size_t)mt2 * 16 + kt3) * 32 + g3 * 16 + m2) * 16 + s3] = (__bf16)h;
    }
    __syncthreads();  // hsw fully written before next step's reads
  }
}

// ---------------------------------------------------------------------------
// Dense + softmax: [32768,512] @ [512,32(pad of 17)] then row softmax(17).
// grid = 256 blocks x 256 threads (8 waves); wave -> one 16-row M-tile.
// ---------------------------------------------------------------------------
__global__ void __launch_bounds__(256)
dense_softmax_kernel(const __bf16* __restrict__ hpack, const __bf16* __restrict__ Wdp,
                     const float* __restrict__ bd, float* __restrict__ out) {
  const int tid = threadIdx.x;
  const int wave = tid >> 5, lane = tid & 31;
  const int mtile = blockIdx.x * 8 + wave;
  const v8f vz = {0.f, 0.f, 0.f, 0.f, 0.f, 0.f, 0.f, 0.f};
  v8f a0 = vz, a1 = vz;

#pragma unroll
  for (int kt = 0; kt < 16; ++kt) {
    v16bf a  = *(const v16bf*)(hpack + (((size_t)mtile * 16 + kt) * 32 + lane) * 16);
    v16bf b0 = *(const v16bf*)(Wdp + (((size_t)kt * 2 + 0) * 32 + lane) * 16);
    v16bf b1 = *(const v16bf*)(Wdp + (((size_t)kt * 2 + 1) * 32 + lane) * 16);
    a0 = WMMA_BF16(a, b0, a0);
    a1 = WMMA_BF16(a, b1, a1);
  }

  __shared__ float tile[8][16][32];  // 16 KB
  int col = lane & 15, ro = (lane >> 4) * 8;
#pragma unroll
  for (int r = 0; r < 8; ++r) {
    tile[wave][ro + r][col]      = a0[r];
    tile[wave][ro + r][16 + col] = a1[r];
  }
  __syncthreads();

  if (tid < 128) {
    int w = tid >> 4, r = tid & 15;
    int rowM = (blockIdx.x * 8 + w) * 16 + r;
    float v[NTAGS];
    float mx = -3.4e38f;
#pragma unroll
    for (int j = 0; j < NTAGS; ++j) {
      v[j] = tile[w][r][j] + bd[j];
      mx = fmaxf(mx, v[j]);
    }
    float ssum = 0.f;
#pragma unroll
    for (int j = 0; j < NTAGS; ++j) {
      v[j] = __expf(v[j] - mx);
      ssum += v[j];
    }
    float inv = 1.f / ssum;
#pragma unroll
    for (int j = 0; j < NTAGS; ++j) out[(size_t)rowM * NTAGS + j] = v[j] * inv;
  }
}

// ---------------------------------------------------------------------------
// Host launch
// ---------------------------------------------------------------------------
extern "C" void kernel_launch(void* const* d_in, const int* in_sizes, int n_in,
                              void* d_out, int out_size, void* d_ws, size_t ws_size,
                              hipStream_t stream) {
  (void)in_sizes; (void)n_in; (void)out_size; (void)ws_size;

  const int*   tokens = (const int*)d_in[0];
  const float* emb    = (const float*)d_in[1];
  const float* Wf     = (const float*)d_in[2];
  const float* Uf     = (const float*)d_in[3];
  const float* bf_    = (const float*)d_in[4];
  const float* Wb     = (const float*)d_in[5];
  const float* Ub     = (const float*)d_in[6];
  const float* bb_    = (const float*)d_in[7];
  const float* Wd     = (const float*)d_in[8];
  const float* bd     = (const float*)d_in[9];
  float* out = (float*)d_out;

  // workspace layout (bytes)
  constexpr size_t SZ_XA = (size_t)MT * 8 * 32 * 16 * 2;     //  16 MB  x A-frags
  constexpr size_t SZ_WU = (size_t)8 * 64 * 32 * 16 * 2;     // 512 KB  each
  constexpr size_t SZ_WD = (size_t)16 * 2 * 32 * 16 * 2;     //  32 KB
  constexpr size_t SZ_XZ = (size_t)ROWS * G4H * 4;           // 128 MB  each
  constexpr size_t o_xa  = 0;
  constexpr size_t o_wf  = o_xa + SZ_XA;
  constexpr size_t o_wb  = o_wf + SZ_WU;
  constexpr size_t o_uf  = o_wb + SZ_WU;
  constexpr size_t o_ub  = o_uf + SZ_WU;
  constexpr size_t o_wd  = o_ub + SZ_WU;
  constexpr size_t o_xzf = o_wd + SZ_WD;
  constexpr size_t o_xzb = o_xzf + SZ_XZ;
  constexpr size_t o_hp  = o_xzb + SZ_XZ;

  char* ws = (char*)d_ws;
  __bf16* xa   = (__bf16*)(ws + o_xa);
  __bf16* wfp  = (__bf16*)(ws + o_wf);
  __bf16* wbp  = (__bf16*)(ws + o_wb);
  __bf16* ufp  = (__bf16*)(ws + o_uf);
  __bf16* ubp  = (__bf16*)(ws + o_ub);
  __bf16* wdp  = (__bf16*)(ws + o_wd);
  float*  xzf  = (float*)(ws + o_xzf);
  float*  xzb  = (float*)(ws + o_xzb);
  __bf16* hp   = (__bf16*)(ws + o_hp);

  // 1. pack weights into B-fragment layouts
  pack_b_kernel<<<1024, 256, 0, stream>>>(Wf, wfp, EE, G4H, 8, 64);
  pack_b_kernel<<<1024, 256, 0, stream>>>(Wb, wbp, EE, G4H, 8, 64);
  pack_b_kernel<<<1024, 256, 0, stream>>>(Uf, ufp, HH, G4H, 8, 64);
  pack_b_kernel<<<1024, 256, 0, stream>>>(Ub, ubp, HH, G4H, 8, 64);
  pack_b_kernel<<<64,   256, 0, stream>>>(Wd, wdp, 2 * HH, NTAGS, 16, 2);

  // 2. embedding gather -> A-fragment bf16
  gather_pack_kernel<<<ROWS, 256, 0, stream>>>(tokens, emb, xa);

  // 3. input projections (fwd, bwd)
  proj_gemm_kernel<<<MT, 256, 0, stream>>>(xa, wfp, xzf);
  proj_gemm_kernel<<<MT, 256, 0, stream>>>(xa, wbp, xzb);

  // 4. bidirectional recurrent scan (16 waves per workgroup)
  scan_kernel<<<dim3(BB / 16, 2), 512, 0, stream>>>(xzf, xzb, bf_, bb_, ufp, ubp, hp);

  // 5. dense + softmax
  dense_softmax_kernel<<<MT / 8, 256, 0, stream>>>(hp, wdp, bd, out);
}
